// TensorNetInteraction_23596550324528
// MI455X (gfx1250) — matrix operand order repaired and checked by reading, not compiled, not run on hardware
//
#include <hip/hip_runtime.h>

typedef _Float16 half_t;
typedef __attribute__((ext_vector_type(16))) _Float16 v16h;
typedef __attribute__((ext_vector_type(8)))  _Float16 v8h;
typedef __attribute__((ext_vector_type(8)))  float    v8f;

#define CUTOFF_ 5.0f
#define PI_F 3.14159265358979f

__device__ __forceinline__ float silu_f(float x) { return x / (1.0f + __expf(-x)); }

__device__ __forceinline__ v8f wmma_f16(v16h a, v16h b, v8f c) {
  // D = A(16x32 f16) * B(32x16 f16) + C(16x16 f32)
  return __builtin_amdgcn_wmma_f32_16x16x32_f16(
      /*neg_a=*/false, a, /*neg_b=*/false, b,
      /*c_mod=*/(short)0, c, /*reuse_a=*/false, /*reuse_b=*/false);
}

// ---- fragment loaders (layouts per CDNA5 ISA 7.12.2, wave32) ----
// All loads are unconditional, 16-byte vector loads (ds_load_b128 / global_load_b128).

// A-matrix 16x32 f16 chunk from row-major MxK half array (ld = ldk, mult of 8):
// lane l: row m = mbase + (l&15); regs 0-3 hold K = kk + khalf + 0..7,
// regs 4-7 hold K = kk + 16 + khalf + 0..7, khalf = (l<16)?0:8.
__device__ __forceinline__ v16h load_afrag(const half_t* Wb, int mbase, int ldk,
                                           int kk, int lane) {
  int m = mbase + (lane & 15);
  int khalf = (lane < 16) ? 0 : 8;
  const half_t* p = Wb + m * ldk + kk + khalf;
  union { v16h v; v8h h[2]; } u;
  u.h[0] = *(const v8h*)(p);
  u.h[1] = *(const v8h*)(p + 16);
  return u.v;
}

// Same A layout from row-major f32 (two 8-float runs, converted on load)
__device__ __forceinline__ v16h load_afrag_f32(const float* Xp, int ldk, int kk, int lane) {
  int m = lane & 15;
  int khalf = (lane < 16) ? 0 : 8;
  const float* p = Xp + m * ldk + kk + khalf;
  const float4 lo0 = *(const float4*)(p);
  const float4 lo1 = *(const float4*)(p + 4);
  const float4 hi0 = *(const float4*)(p + 16);
  const float4 hi1 = *(const float4*)(p + 20);
  v16h a;
  a[0] = (half_t)lo0.x;  a[1] = (half_t)lo0.y;  a[2] = (half_t)lo0.z;  a[3] = (half_t)lo0.w;
  a[4] = (half_t)lo1.x;  a[5] = (half_t)lo1.y;  a[6] = (half_t)lo1.z;  a[7] = (half_t)lo1.w;
  a[8] = (half_t)hi0.x;  a[9] = (half_t)hi0.y;  a[10] = (half_t)hi0.z; a[11] = (half_t)hi0.w;
  a[12] = (half_t)hi1.x; a[13] = (half_t)hi1.y; a[14] = (half_t)hi1.z; a[15] = (half_t)hi1.w;
  return a;
}

// B-matrix 32x16 chunk from a TRANSPOSED staging array Bt[16 cols][ldk]:
// lane l: col n = l&15; elements 0-7 -> K = kk + khalf + 0..7, 8-15 -> K = kk + 16 + khalf + 0..7, khalf = (l<16)?0:8
__device__ __forceinline__ v16h load_bfrag_t(const half_t* Bt, int ldk, int kk, int lane) {
  int n = lane & 15;
  int kofs = kk + ((lane < 16) ? 0 : 8);
  const half_t* p = Bt + n * ldk + kofs;
  union { v16h v; v8h h[2]; } u;
  u.h[0] = *(const v8h*)(p);
  u.h[1] = *(const v8h*)(p + 16);
  return u.v;
}

// B = W^T where W is row-major [Nout x K] (ld = ldk): B[k][n] = W[n][k]
// -> two 8-half runs per lane (K = kk + khalf + 0..7 and kk + 16 + khalf + 0..7).
__device__ __forceinline__ v16h load_bfrag_wT(const half_t* W, int nbase, int ldk,
                                              int kk, int lane) {
  int n = nbase + (lane & 15);
  int kofs = kk + ((lane < 16) ? 0 : 8);
  const half_t* p = W + n * ldk + kofs;
  union { v16h v; v8h h[2]; } u;
  u.h[0] = *(const v8h*)(p);
  u.h[1] = *(const v8h*)(p + 16);
  return u.v;
}

// D store (16x16 f32): lane l holds col c=l&15, rows v + (l<16?0:8); keep cols < 9
__device__ __forceinline__ void store_d_cols9(v8f acc, float* dst, int gbase, int lane) {
  int c = lane & 15;
  int ro = (lane < 16) ? 0 : 8;
  if (c < 9) {
#pragma unroll
    for (int v = 0; v < 8; ++v)
      dst[(gbase + v + ro) * 9 + c] = acc[v];
  }
}

// ---- kernel 0: convert all weights to f16 (packed blob in ws) ----
// layout (halves): Wt0..Wt5 at i*4096, Ws1 @24576 [64x32], Ws2 @26624 [128x64], Ws3 @34816 [192x128]
__global__ void k_convert(const float* Wt0, const float* Wt1, const float* Wt2,
                          const float* Wt3, const float* Wt4, const float* Wt5,
                          const float* Ws1, const float* Ws2, const float* Ws3,
                          half_t* Wh) {
  int total = 6 * 4096 + 2048 + 8192 + 24576;  // 59392
  for (int i = blockIdx.x * blockDim.x + threadIdx.x; i < total;
       i += gridDim.x * blockDim.x) {
    float v;
    if (i < 4096)       v = Wt0[i];
    else if (i < 8192)  v = Wt1[i - 4096];
    else if (i < 12288) v = Wt2[i - 8192];
    else if (i < 16384) v = Wt3[i - 12288];
    else if (i < 20480) v = Wt4[i - 16384];
    else if (i < 24576) v = Wt5[i - 20480];
    else if (i < 26624) v = Ws1[i - 24576];
    else if (i < 34816) v = Ws2[i - 26624];
    else                v = Ws3[i - 34816];
    Wh[i] = (half_t)v;
  }
}

__global__ void k_zero(float* p, int n) {
  int i = blockIdx.x * blockDim.x + threadIdx.x;
  if (i < n) p[i] = 0.0f;
}

// ---- kernel 1: per-node normalize + decompose + channel-linear (Wt0/1/2) ----
__global__ void __launch_bounds__(128) k_node_prep(
    const float* __restrict__ X, const half_t* __restrict__ Wh,
    float* __restrict__ Xn, float* __restrict__ Iout,
    float* __restrict__ Aout, float* __restrict__ Sout) {
  // transposed staging: [tensor][component col 0..15][h 0..63 (+pad)]
  __shared__ __align__(16) half_t sPt[3][16][72];
  int node = blockIdx.x, tid = threadIdx.x;
  const float* xp = X + (size_t)node * 576;
  if (tid < 64) {
    float x[9], t = 0.f;
#pragma unroll
    for (int c = 0; c < 9; ++c) { x[c] = xp[tid * 9 + c]; t += x[c] * x[c]; }
    float inv = 1.0f / (t + 1.0f);
    float xn[9];
#pragma unroll
    for (int c = 0; c < 9; ++c) {
      xn[c] = x[c] * inv;
      Xn[(size_t)node * 576 + tid * 9 + c] = xn[c];
    }
    float tr = (xn[0] + xn[4] + xn[8]) * (1.0f / 3.0f);
#pragma unroll
    for (int i = 0; i < 3; ++i)
#pragma unroll
      for (int j = 0; j < 3; ++j) {
        float a = 0.5f * (xn[i * 3 + j] - xn[j * 3 + i]);
        float s = 0.5f * (xn[i * 3 + j] + xn[j * 3 + i]) - ((i == j) ? tr : 0.f);
        sPt[0][i * 3 + j][tid] = (half_t)((i == j) ? tr : 0.f);
        sPt[1][i * 3 + j][tid] = (half_t)a;
        sPt[2][i * 3 + j][tid] = (half_t)s;
      }
#pragma unroll
    for (int c = 9; c < 16; ++c) {  // zero the unused B columns
      sPt[0][c][tid] = (half_t)0.0f;
      sPt[1][c][tid] = (half_t)0.0f;
      sPt[2][c][tid] = (half_t)0.0f;
    }
  }
  __syncthreads();
  int wave = tid >> 5, lane = tid & 31, mt = wave;  // 4 waves = 4 M-tiles of g
  float* outs[3] = {Iout, Aout, Sout};
#pragma unroll
  for (int t = 0; t < 3; ++t) {
    v8f acc = {};
#pragma unroll
    for (int kk = 0; kk < 64; kk += 32) {
      v16h a = load_afrag(Wh + t * 4096, mt * 16, 64, kk, lane);   // Wt[g,h]
      v16h b = load_bfrag_t(&sPt[t][0][0], 72, kk, lane);          // val^T[c][h]
      acc = wmma_f16(a, b, acc);
    }
    store_d_cols9(acc, outs[t] + (size_t)node * 576, mt * 16, lane);
  }
}

// ---- kernel 2: fused edge MLP (3x WMMA GEMM) + gather/scatter message passing ----
__global__ void __launch_bounds__(128) k_edge(
    const float* __restrict__ eattr, const float* __restrict__ ew,
    const int* __restrict__ eidx,
    const float* __restrict__ bs1, const float* __restrict__ bs2,
    const float* __restrict__ bs3, const half_t* __restrict__ Wh,
    const float* __restrict__ Ipost, const float* __restrict__ Apost,
    const float* __restrict__ Spost, float* __restrict__ msg, int E_) {
  __shared__ __align__(16) half_t hst1[4][16][72];    // h1: 16 edges x 64 f16 (row = edge)
  __shared__ __align__(16) half_t hst2[4][16][136];   // h2: 16 edges x 128 f16
  __shared__ __align__(16) float  eas[4][16][192];    // h3*C: 16 edges x 192 f32
  __shared__ float ccs[4][16];                        // cosine cutoff per edge
  int tid = threadIdx.x, wave = tid >> 5, lane = tid & 31;
  int ebase = blockIdx.x * 64 + wave * 16;
  const half_t* W1h = Wh + 24576;   // [64 x 32]
  const half_t* W2h = Wh + 26624;   // [128 x 64]
  const half_t* W3h = Wh + 34816;   // [192 x 128]

  if (lane < 16) {
    float w = ew[ebase + lane];
    float c = 0.5f * (__cosf(PI_F * w * (1.0f / CUTOFF_)) + 1.0f);
    ccs[wave][lane] = (w < CUTOFF_) ? c : 0.0f;
  }
  // GEMM1: [16x32] @ Ws1^T[32x64]
  v16h a0 = load_afrag_f32(eattr + (size_t)ebase * 32, 32, 0, lane);
#pragma unroll
  for (int nt = 0; nt < 4; ++nt) {
    v16h b = load_bfrag_wT(W1h, nt * 16, 32, 0, lane);
    v8f acc = {};
    acc = wmma_f16(a0, b, acc);
    int c = lane & 15, col = nt * 16 + c, ro = (lane < 16) ? 0 : 8;
    float bias = bs1[col];
#pragma unroll
    for (int v = 0; v < 8; ++v)
      hst1[wave][v + ro][col] = (half_t)silu_f(acc[v] + bias);
  }
  __syncthreads();
  // GEMM2: [16x64] @ Ws2^T[64x128]
#pragma unroll
  for (int nt = 0; nt < 8; ++nt) {
    v8f acc = {};
#pragma unroll
    for (int kk = 0; kk < 64; kk += 32) {
      v16h a = load_afrag(&hst1[wave][0][0], 0, 72, kk, lane);
      v16h b = load_bfrag_wT(W2h, nt * 16, 64, kk, lane);
      acc = wmma_f16(a, b, acc);
    }
    int c = lane & 15, col = nt * 16 + c, ro = (lane < 16) ? 0 : 8;
    float bias = bs2[col];
#pragma unroll
    for (int v = 0; v < 8; ++v)
      hst2[wave][v + ro][col] = (half_t)silu_f(acc[v] + bias);
  }
  __syncthreads();
  // GEMM3: [16x128] @ Ws3^T[128x192], apply silu + cutoff
#pragma unroll
  for (int nt = 0; nt < 12; ++nt) {
    v8f acc = {};
#pragma unroll
    for (int kk = 0; kk < 128; kk += 32) {
      v16h a = load_afrag(&hst2[wave][0][0], 0, 136, kk, lane);
      v16h b = load_bfrag_wT(W3h, nt * 16, 128, kk, lane);
      acc = wmma_f16(a, b, acc);
    }
    int c = lane & 15, col = nt * 16 + c, ro = (lane < 16) ? 0 : 8;
    float bias = bs3[col];
#pragma unroll
    for (int v = 0; v < 8; ++v)
      eas[wave][v + ro][col] = silu_f(acc[v] + bias) * ccs[wave][v + ro];
  }
  __syncthreads();
  // scatter: msg[dst] += ea0*I[src] + ea1*A[src] + ea2*S[src]  (L2-resident gathers)
  for (int i = 0; i < 16; ++i) {
    int e0 = ebase + i;
    int dst = eidx[e0];        // edge_index[0]
    int src = eidx[E_ + e0];   // edge_index[1]
    const float* ip = Ipost + (size_t)src * 576;
    const float* ap = Apost + (size_t)src * 576;
    const float* sp = Spost + (size_t)src * 576;
    float* mp = msg + (size_t)dst * 576;
    if (i + 1 < 16) {  // prefetch next edge's source rows (global_prefetch_b8)
      int s2 = eidx[E_ + e0 + 1];
      __builtin_prefetch(Ipost + (size_t)s2 * 576 + lane * 18, 0, 1);
      __builtin_prefetch(Apost + (size_t)s2 * 576 + lane * 18, 0, 1);
      __builtin_prefetch(Spost + (size_t)s2 * 576 + lane * 18, 0, 1);
    }
    for (int idx = lane; idx < 576; idx += 32) {
      int h = idx / 9;
      float v = eas[wave][i][h * 3 + 0] * ip[idx]
              + eas[wave][i][h * 3 + 1] * ap[idx]
              + eas[wave][i][h * 3 + 2] * sp[idx];
      atomicAdd(mp + idx, v);
    }
  }
}

// ---- kernel 3: O(3) product, decompose+norm, channel-linear (Wt3/4/5), output ----
__global__ void __launch_bounds__(128) k_final(
    const float* __restrict__ msg, const float* __restrict__ Ipost,
    const float* __restrict__ Apost, const float* __restrict__ Spost,
    const float* __restrict__ Xn, const float* __restrict__ q,
    const half_t* __restrict__ Wh, float* __restrict__ out) {
  __shared__ __align__(16) half_t sPt[3][16][72];  // transposed staging
  __shared__ float dx[64][9];
  int node = blockIdx.x, tid = threadIdx.x;
  float qf = 1.0f + 0.1f * q[node];
  if (tid < 64) {
    size_t base = (size_t)node * 576 + tid * 9;
    float m9[9], y9[9];
#pragma unroll
    for (int c = 0; c < 9; ++c) {
      m9[c] = msg[base + c];
      y9[c] = Ipost[base + c] + Apost[base + c] + Spost[base + c];  // Y = I+A+S
    }
    float T[9];
#pragma unroll
    for (int i = 0; i < 3; ++i)
#pragma unroll
      for (int j = 0; j < 3; ++j) {
        float acc = 0.f;
#pragma unroll
        for (int k = 0; k < 3; ++k)
          acc += m9[i * 3 + k] * y9[k * 3 + j] + y9[i * 3 + k] * m9[k * 3 + j];
        T[i * 3 + j] = qf * acc;
      }
    float tn = 0.f;
#pragma unroll
    for (int c = 0; c < 9; ++c) tn += T[c] * T[c];
    float inv = 1.0f / (tn + 1.0f);
    float tr = (T[0] + T[4] + T[8]) * (1.0f / 3.0f);
#pragma unroll
    for (int i = 0; i < 3; ++i)
#pragma unroll
      for (int j = 0; j < 3; ++j) {
        float a = 0.5f * (T[i * 3 + j] - T[j * 3 + i]);
        float s = 0.5f * (T[i * 3 + j] + T[j * 3 + i]) - ((i == j) ? tr : 0.f);
        sPt[0][i * 3 + j][tid] = (half_t)(((i == j) ? tr : 0.f) * inv);
        sPt[1][i * 3 + j][tid] = (half_t)(a * inv);
        sPt[2][i * 3 + j][tid] = (half_t)(s * inv);
      }
#pragma unroll
    for (int c = 9; c < 16; ++c) {
      sPt[0][c][tid] = (half_t)0.0f;
      sPt[1][c][tid] = (half_t)0.0f;
      sPt[2][c][tid] = (half_t)0.0f;
    }
  }
  __syncthreads();
  int wave = tid >> 5, lane = tid & 31, mt = wave;
  {
    v8f acc = {};
#pragma unroll
    for (int t = 0; t < 3; ++t)
#pragma unroll
      for (int kk = 0; kk < 64; kk += 32) {
        v16h a = load_afrag(Wh + (3 + t) * 4096, mt * 16, 64, kk, lane);
        v16h b = load_bfrag_t(&sPt[t][0][0], 72, kk, lane);
        acc = wmma_f16(a, b, acc);  // dX = sum of three chan-linears
      }
    int c = lane & 15, ro = (lane < 16) ? 0 : 8;
    if (c < 9)
#pragma unroll
      for (int v = 0; v < 8; ++v)
        dx[mt * 16 + v + ro][c] = acc[v];
  }
  __syncthreads();
  if (tid < 64) {
    int g = tid;
    float d9[9], xn9[9];
#pragma unroll
    for (int c = 0; c < 9; ++c) {
      d9[c] = dx[g][c];
      xn9[c] = Xn[(size_t)node * 576 + g * 9 + c];
    }
#pragma unroll
    for (int i = 0; i < 3; ++i)
#pragma unroll
      for (int j = 0; j < 3; ++j) {
        float acc2 = 0.f;
#pragma unroll
        for (int k = 0; k < 3; ++k) acc2 += d9[i * 3 + k] * d9[k * 3 + j];
        out[(size_t)node * 576 + g * 9 + i * 3 + j] =
            xn9[i * 3 + j] + d9[i * 3 + j] + qf * acc2;
      }
  }
}

extern "C" void kernel_launch(void* const* d_in, const int* in_sizes, int n_in,
                              void* d_out, int out_size, void* d_ws, size_t ws_size,
                              hipStream_t stream) {
  const float* X     = (const float*)d_in[0];
  const int*   eidx  = (const int*)d_in[1];
  const float* ew    = (const float*)d_in[2];
  const float* eattr = (const float*)d_in[3];
  const float* q     = (const float*)d_in[4];
  const float* Ws1   = (const float*)d_in[5];
  const float* bs1   = (const float*)d_in[6];
  const float* Ws2   = (const float*)d_in[7];
  const float* bs2   = (const float*)d_in[8];
  const float* Ws3   = (const float*)d_in[9];
  const float* bs3   = (const float*)d_in[10];
  const float* Wt0   = (const float*)d_in[11];
  const float* Wt1   = (const float*)d_in[12];
  const float* Wt2   = (const float*)d_in[13];
  const float* Wt3   = (const float*)d_in[14];
  const float* Wt4   = (const float*)d_in[15];
  const float* Wt5   = (const float*)d_in[16];

  int nN = in_sizes[0] / 576;  // 20000
  int nE = in_sizes[2];        // 320000

  // workspace layout
  half_t* Wh = (half_t*)d_ws;                       // 59392 halves, padded to 128 KB
  float* fb = (float*)((char*)d_ws + 131072);
  size_t stride = (size_t)nN * 576;
  float* Xn    = fb;
  float* Ipost = Xn + stride;
  float* Apost = Ipost + stride;
  float* Spost = Apost + stride;
  float* msgb  = Spost + stride;

  k_convert<<<64, 256, 0, stream>>>(Wt0, Wt1, Wt2, Wt3, Wt4, Wt5, Ws1, Ws2, Ws3, Wh);
  int zb = (int)((stride + 255) / 256);
  k_zero<<<zb, 256, 0, stream>>>(msgb, (int)stride);
  k_node_prep<<<nN, 128, 0, stream>>>(X, Wh, Xn, Ipost, Apost, Spost);
  k_edge<<<nE / 64, 128, 0, stream>>>(eattr, ew, eidx, bs1, bs2, bs3, Wh,
                                      Ipost, Apost, Spost, msgb, nE);
  k_final<<<nN, 128, 0, stream>>>(msgb, Ipost, Apost, Spost, Xn, q, Wh, (float*)d_out);
}
